// NTXentLoss_54004918779943
// MI455X (gfx1250) — compile-verified, matrix operand order checked by reading, not run on hardware
//
#include <hip/hip_runtime.h>

typedef __attribute__((ext_vector_type(16))) _Float16 v16h;
typedef __attribute__((ext_vector_type(8)))  float    v8f;
typedef __attribute__((ext_vector_type(4)))  _Float16 h4;

#define TWO_B  8192
#define HALF_B 4096
#define CDIM   128
#define INV_T  2.0f                    // 1 / TEMPERATURE
#define K2E    2.8853900817779268f     // INV_T * log2(e)  -> exp(x/T) = exp2(x*K2E)
#define LN2    0.6931471805599453f

// Raw hardware exp2/log2 (v_exp_f32 / v_log_f32). Safe here: exp2 arguments
// are bounded by |sim * K2E| <= ~2.9 (cosine similarities in [-1,1]); log
// argument is a large positive sum. No OCML denorm range-fixup needed.
__device__ __forceinline__ float fast_exp2(float x) { return __builtin_amdgcn_exp2f(x); }
__device__ __forceinline__ float fast_log2(float x) { return __builtin_amdgcn_logf(x); }

union FragU { v16h h; uint4 q[2]; };

// ---------------------------------------------------------------------------
// Kernel 1: L2-normalize rows, write f16 reps. One wave per row.
// ---------------------------------------------------------------------------
__global__ __launch_bounds__(256) void nrm_kernel(const float* __restrict__ zi,
                                                  const float* __restrict__ zj,
                                                  _Float16* __restrict__ reps) {
    int wave = threadIdx.x >> 5;
    int lane = threadIdx.x & 31;
    int row  = blockIdx.x * 8 + wave;
    const float* src = (row < HALF_B) ? (zi + (size_t)row * CDIM)
                                      : (zj + (size_t)(row - HALF_B) * CDIM);
    float4 v = ((const float4*)src)[lane];
    float ss = v.x * v.x + v.y * v.y + v.z * v.z + v.w * v.w;
    #pragma unroll
    for (int m = 16; m >= 1; m >>= 1) ss += __shfl_xor(ss, m, 32);
    float rn = rsqrtf(ss);
    h4 o;
    o[0] = (_Float16)(v.x * rn);
    o[1] = (_Float16)(v.y * rn);
    o[2] = (_Float16)(v.z * rn);
    o[3] = (_Float16)(v.w * rn);
    *(h4*)(reps + (size_t)row * CDIM + lane * 4) = o;
}

// ---------------------------------------------------------------------------
// Kernel 2: per-row positive dot (row R vs row (R+B) mod 2B) and exact
// diagonal self-dot of the f16 rows. One wave per row. Keeps ALL capture
// logic out of the GEMM hot loop.
// ---------------------------------------------------------------------------
__global__ __launch_bounds__(256) void posdiag_kernel(
        const _Float16* __restrict__ reps,
        float* __restrict__ posArr, float* __restrict__ diagArr) {
    int wave = threadIdx.x >> 5;
    int lane = threadIdx.x & 31;
    int row  = blockIdx.x * 8 + wave;
    int prt  = (row + HALF_B) & (TWO_B - 1);
    h4 a = *(const h4*)(reps + (size_t)row * CDIM + lane * 4);
    h4 b = *(const h4*)(reps + (size_t)prt * CDIM + lane * 4);
    float p = 0.0f, d = 0.0f;
    #pragma unroll
    for (int k = 0; k < 4; ++k) {
        float av = (float)a[k], bv = (float)b[k];
        p += av * bv;
        d += av * av;
    }
    #pragma unroll
    for (int m = 16; m >= 1; m >>= 1) {
        p += __shfl_xor(p, m, 32);
        d += __shfl_xor(d, m, 32);
    }
    if (lane == 0) { posArr[row] = p; diagArr[row] = d; }
}

// ---------------------------------------------------------------------------
// Kernel 3: fused sim = reps @ reps^T with branch-free online reduction.
// 256 workgroups; workgroup g owns rows [32g, 32g+32).
// 8 waves: rowTile = wave>>2 (2 tiles of 16 rows), colQ = wave&3.
// Column loop: 64 iterations of 128 columns; each wave does TWO 16x16 tiles
// per iteration = 8 chained v_wmma_f32_16x16x32_f16.
// Accumulates exp(sim/T) over ALL columns; diag/pos fixed up at the end.
// ---------------------------------------------------------------------------
__global__ __launch_bounds__(256) void simclr_wmma_kernel(
        const _Float16* __restrict__ reps,
        const float* __restrict__ posArr, const float* __restrict__ diagArr,
        float* __restrict__ blockLoss) {
    __shared__ _Float16 Apanel[32 * CDIM];      // 8 KB f16 A stripe
    __shared__ float rowSumSh[32];

    int tid  = threadIdx.x;
    int lane = tid & 31;
    int wave = tid >> 5;
    int rowTile = wave >> 2;                    // 0..1
    int colQ    = wave & 3;                     // 0..3
    int rowBase = blockIdx.x * 32;

    // Stage the 32x128 f16 A panel into LDS.
    {
        const uint4* gsrc = (const uint4*)(reps + (size_t)rowBase * CDIM);
        uint4* ldst = (uint4*)Apanel;
        for (int i = tid; i < (32 * CDIM) / 8; i += 256) ldst[i] = gsrc[i];
    }
    if (tid < 32) rowSumSh[tid] = 0.0f;
    __syncthreads();

    int hgrp = lane >> 4;
    int mr   = lane & 15;

    // A fragments (16x128) held in registers for the whole column sweep.
    FragU a[4];
    const _Float16* arow = Apanel + (size_t)(rowTile * 16 + mr) * CDIM;
    #pragma unroll
    for (int k = 0; k < 4; ++k) {
        int K0 = k * 32 + hgrp * 8;
        a[k].q[0] = *(const uint4*)(arow + K0);
        a[k].q[1] = *(const uint4*)(arow + K0 + 16);
    }

    float acc[8];
    #pragma unroll
    for (int p = 0; p < 8; ++p) acc[p] = 0.0f;

    for (int ct = 0; ct < TWO_B / 128; ++ct) {
        int colBase = ct * 128 + colQ * 32;     // this wave: cols [colBase, +32)
        const _Float16* brow0 = reps + (size_t)(colBase + mr) * CDIM;
        const _Float16* brow1 = reps + (size_t)(colBase + 16 + mr) * CDIM;
        FragU b0[4], b1[4];
        #pragma unroll
        for (int k = 0; k < 4; ++k) {
            int K0 = k * 32 + hgrp * 8;
            b0[k].q[0] = *(const uint4*)(brow0 + K0);
            b0[k].q[1] = *(const uint4*)(brow0 + K0 + 16);
            b1[k].q[0] = *(const uint4*)(brow1 + K0);
            b1[k].q[1] = *(const uint4*)(brow1 + K0 + 16);
        }
        v8f c0 = {}, c1 = {};
        #pragma unroll
        for (int k = 0; k < 4; ++k) {
            c0 = __builtin_amdgcn_wmma_f32_16x16x32_f16(
                     false, a[k].h, false, b0[k].h, (short)0, c0, false, false);
            c1 = __builtin_amdgcn_wmma_f32_16x16x32_f16(
                     false, a[k].h, false, b1[k].h, (short)0, c1, false, false);
        }
        // Branch-free consume: sum exp(sim/T) over every column.
        // Bare v_exp_f32, no range fixup (arguments bounded by ~±2.9).
        #pragma unroll
        for (int p = 0; p < 8; ++p)
            acc[p] += fast_exp2(c0[p] * K2E) + fast_exp2(c1[p] * K2E);
    }

    // Reduce each row partial across its 16 lanes, merge the 4 colQ waves.
    #pragma unroll
    for (int p = 0; p < 8; ++p) {
        float s = acc[p];
        #pragma unroll
        for (int m = 8; m >= 1; m >>= 1) s += __shfl_xor(s, m, 32);
        if (mr == 0)
            atomicAdd(&rowSumSh[rowTile * 16 + p + 8 * hgrp], s);
    }
    __syncthreads();

    // Fix-up: remove masked diagonal term, add the prefix positive column.
    // loss_i = log(S_i) - pos_i/T.
    if (tid < 32) {
        int R = rowBase + tid;
        float pos  = posArr[R];
        float diag = diagArr[R];
        float S = rowSumSh[tid] - fast_exp2(diag * K2E) + fast_exp2(pos * K2E);
        rowSumSh[tid] = fast_log2(S) * LN2 - pos * INV_T;
    }
    __syncthreads();
    if (tid == 0) {
        float t = 0.0f;
        for (int r = 0; r < 32; ++r) t += rowSumSh[r];
        blockLoss[blockIdx.x] = t;   // plain store: deterministic per replay
    }
}

// ---------------------------------------------------------------------------
// Kernel 4: sum the 256 per-block losses, write the mean (plain store).
// ---------------------------------------------------------------------------
__global__ __launch_bounds__(256) void finalize_kernel(
        const float* __restrict__ blockLoss, float* __restrict__ out) {
    int lane = threadIdx.x & 31;
    int wave = threadIdx.x >> 5;
    __shared__ float sh[8];
    float v = blockLoss[threadIdx.x];
    #pragma unroll
    for (int m = 16; m >= 1; m >>= 1) v += __shfl_xor(v, m, 32);
    if (lane == 0) sh[wave] = v;
    __syncthreads();
    if (threadIdx.x == 0) {
        float t = 0.0f;
        for (int i = 0; i < 8; ++i) t += sh[i];
        out[0] = t / (float)TWO_B;
    }
}

extern "C" void kernel_launch(void* const* d_in, const int* in_sizes, int n_in,
                              void* d_out, int out_size, void* d_ws, size_t ws_size,
                              hipStream_t stream) {
    (void)in_sizes; (void)n_in; (void)out_size; (void)ws_size;
    const float* zi = (const float*)d_in[0];
    const float* zj = (const float*)d_in[1];

    char* ws = (char*)d_ws;
    _Float16* reps    = (_Float16*)ws;                       // 2 MB
    float* posArr     = (float*)(ws + (size_t)TWO_B * CDIM * sizeof(_Float16));
    float* diagArr    = posArr + TWO_B;                      // 32 KB each
    float* blockLoss  = diagArr + TWO_B;                     // 1 KB

    nrm_kernel<<<TWO_B / 8, 256, 0, stream>>>(zi, zj, reps);
    posdiag_kernel<<<TWO_B / 8, 256, 0, stream>>>(reps, posArr, diagArr);
    simclr_wmma_kernel<<<TWO_B / 32, 256, 0, stream>>>(reps, posArr, diagArr, blockLoss);
    finalize_kernel<<<1, 256, 0, stream>>>(blockLoss, (float*)d_out);
}
